// SimpleRNN_20736102105459
// MI455X (gfx1250) — matrix operand order repaired and checked by reading, not compiled, not run on hardware
//
#include <hip/hip_runtime.h>

// ---------------------------------------------------------------------------
// BiLSTM tagger for MI455X (gfx1250).  All GEMMs run on bf16 WMMA with f32
// accumulate (v_wmma_f32_16x16x32_bf16).  h/c recurrent state stays f32.
// Weights stay in native [N][K] layout (already B-fragment friendly).
// GEMM uses double-buffered LDS (one barrier per K-chunk); staging uses
// GLOBAL_LOAD_ASYNC_TO_LDS_B128 (ASYNCcnt) so the copy bypasses VGPRs and
// overlaps the WMMA block without a mid-iteration s_wait_loadcnt stall.
// ---------------------------------------------------------------------------

typedef unsigned short u16;
typedef __attribute__((ext_vector_type(16))) __bf16 v16bf;
typedef __attribute__((ext_vector_type(8)))  float  v8f;
typedef __attribute__((ext_vector_type(8)))  u16    v8us;
typedef __attribute__((ext_vector_type(4)))  int    v4i;

union FragU { v16bf v; v8us h[2]; };

#define B_   64
#define S_   128
#define W_   16
#define E_   256
#define H_   512
#define NC_  20
#define BS_  (B_*S_)        // 8192 word rows
#define NBS_ (BS_*W_)       // 131072 char rows
#define ROWP 40             // padded LDS row stride (u16 units): 80B, bank-spread

#if __has_builtin(__builtin_amdgcn_global_load_async_to_lds_b128)
#define USE_ASYNC_LDS 1
#else
#define USE_ASYNC_LDS 0
#endif
#define AS1 __attribute__((address_space(1)))
#define AS3 __attribute__((address_space(3)))
typedef AS1 v4i* gptr128;   // global 16B-vector pointer (builtin param type)
typedef AS3 v4i* lptr128;   // LDS    16B-vector pointer (builtin param type)

__device__ __forceinline__ void wait_async_zero() {
#if USE_ASYNC_LDS
#if __has_builtin(__builtin_amdgcn_s_wait_asynccnt)
    __builtin_amdgcn_s_wait_asynccnt(0);
#else
    asm volatile("s_wait_asynccnt 0x0" ::: "memory");
#endif
#endif
}

__device__ __forceinline__ u16 f2bf(float f) {
    unsigned u = __float_as_uint(f);
    u += 0x7FFFu + ((u >> 16) & 1u);           // round-to-nearest-even
    return (u16)(u >> 16);
}
__device__ __forceinline__ float sigm(float x) { return 1.0f / (1.0f + __expf(-x)); }
__device__ __forceinline__ v8f v8f_zero() {
    v8f z;
#pragma unroll
    for (int i = 0; i < 8; ++i) z[i] = 0.0f;
    return z;
}

// ---------------------------------------------------------------------------
// Fused dual-source GEMM:
//   C[M][N] (+)= A0[M][K0] * B0[N][K0]^T  (+ A1[M][K1] * B1[N][K1]^T) (+ bias)
// A* bf16 activations (row-major).  B* bf16 weights, native [N][K] layout.
// Block tile 64x128, 8 waves as 2(M) x 4(N), each wave a 32x32 tile
// (2 A-frags x 2 B-frags -> 4 WMMA acc tiles).
// M % 64 == 0, N % 128 == 0, K % 32 == 0 (guaranteed by problem dims).
// ---------------------------------------------------------------------------
__global__ __launch_bounds__(256)
void gemm2_bf16_wmma(const u16* __restrict__ A0, int lda0, const u16* __restrict__ B0, int K0,
                     const u16* __restrict__ A1, int lda1, const u16* __restrict__ B1, int K1,
                     float* __restrict__ C, int ldc,
                     const float* __restrict__ bias, int accumulate)
{
    __shared__ alignas(16) u16 sA[2][64  * ROWP];   // [buf][row][k]  ~10 KB
    __shared__ alignas(16) u16 sB[2][128 * ROWP];   // [buf][col][k]  ~20 KB

    const int tid  = threadIdx.x;
    const int lane = tid & 31;
    const int wave = tid >> 5;
    const int wm   = wave >> 2;                  // 0..1 (M sub-tile, 32 rows)
    const int wn   = wave & 3;                   // 0..3 (N sub-tile, 32 cols)
    const int row0 = blockIdx.y * 64;
    const int col0 = blockIdx.x * 128;
    const int l15  = lane & 15;
    const int hi   = lane >> 4;

    v8f acc[2][2];
    if (accumulate) {
#pragma unroll
        for (int s2 = 0; s2 < 2; ++s2)
#pragma unroll
            for (int j = 0; j < 2; ++j) {
                int cc = col0 + wn * 32 + j * 16 + l15;
                int cr = row0 + wm * 32 + s2 * 16 + hi * 8;
#pragma unroll
                for (int v = 0; v < 8; ++v)
                    acc[s2][j][v] = C[(size_t)(cr + v) * ldc + cc];
            }
    } else {
#pragma unroll
        for (int s2 = 0; s2 < 2; ++s2)
#pragma unroll
            for (int j = 0; j < 2; ++j) acc[s2][j] = v8f_zero();
    }

    const int ar   = tid >> 2;                   // 0..63   A loader row
    const int aoff = (tid & 3) * 8;              // 0/8/16/24
    const int bn   = tid >> 1;                   // 0..127  B loader col (n)
    const int boff = (tid & 1) * 16;             // 0/16

    const u16* Aa[2] = { A0, A1 };
    const u16* Ba[2] = { B0, B1 };
    const int  lA[2] = { lda0, lda1 };
    const int  Ks[2] = { K0, K1 };

    const int nch0 = K0 / 32;
    const int nch  = nch0 + (A1 ? K1 / 32 : 0);

    // stage chunk c (flattened across the two source pairs) into LDS buffer b
    auto stage = [&](int c, int b) {
        const int p   = (c >= nch0) ? 1 : 0;
        const int k0  = (p ? (c - nch0) : c) * 32;
        const int lda = lA[p], K = Ks[p];
        const u16* aptr = &Aa[p][(size_t)(row0 + ar) * lda + aoff + k0];
        const u16* bptr = &Ba[p][(size_t)(col0 + bn) * K   + boff + k0];
#if USE_ASYNC_LDS
        // direct global->LDS DMA, no VGPR round-trip, tracked by ASYNCcnt
        __builtin_amdgcn_global_load_async_to_lds_b128(
            (gptr128)aptr, (lptr128)&sA[b][ar * ROWP + aoff], 0, 0);
        __builtin_amdgcn_global_load_async_to_lds_b128(
            (gptr128)bptr, (lptr128)&sB[b][bn * ROWP + boff], 0, 0);
        __builtin_amdgcn_global_load_async_to_lds_b128(
            (gptr128)(bptr + 8), (lptr128)&sB[b][bn * ROWP + boff + 8], 0, 0);
#else
        *(v8us*)&sA[b][ar * ROWP + aoff] = *(const v8us*)aptr;
        v8us b0v = *(const v8us*)bptr;
        v8us b1v = *(const v8us*)(bptr + 8);
        *(v8us*)&sB[b][bn * ROWP + boff]     = b0v;
        *(v8us*)&sB[b][bn * ROWP + boff + 8] = b1v;
#endif
        if (k0 + 32 < K) {                       // prefetch next chunk of pair
            __builtin_prefetch(aptr + 32, 0, 1);
            __builtin_prefetch(bptr + 32, 0, 1);
        }
    };

    int buf = 0;
    stage(0, 0);
    for (int c = 0; c < nch; ++c) {
        wait_async_zero();                       // our async copies into 'buf' done
        __syncthreads();                         // buf visible to all; buf^1 free
        if (c + 1 < nch) stage(c + 1, buf ^ 1);  // fire-and-forget, overlaps WMMAs

        FragU af[2], bfr[2];
#pragma unroll
        for (int s2 = 0; s2 < 2; ++s2) {
            // A fragment: lane l -> row (l&15), K = (l>=16)*16 + 0..15 contiguous
            const u16* ap = &sA[buf][(wm * 32 + s2 * 16 + l15) * ROWP + hi * 16];
            af[s2].h[0] = *(const v8us*)ap;
            af[s2].h[1] = *(const v8us*)(ap + 8);
        }
#pragma unroll
        for (int j = 0; j < 2; ++j) {
            // B fragment: lane l -> col (l&15), K = (l>=16)*16 + 0..15 contiguous
            const u16* bp = &sB[buf][(wn * 32 + j * 16 + l15) * ROWP + hi * 16];
            bfr[j].h[0] = *(const v8us*)bp;
            bfr[j].h[1] = *(const v8us*)(bp + 8);
        }
#pragma unroll
        for (int s2 = 0; s2 < 2; ++s2)
#pragma unroll
            for (int j = 0; j < 2; ++j)
                acc[s2][j] = __builtin_amdgcn_wmma_f32_16x16x32_bf16(
                    false, af[s2].v, false, bfr[j].v, (short)0, acc[s2][j],
                    false, false);
        buf ^= 1;
    }

#pragma unroll
    for (int s2 = 0; s2 < 2; ++s2)
#pragma unroll
        for (int j = 0; j < 2; ++j) {
            int cc = col0 + wn * 32 + j * 16 + l15;
            float bv = bias ? bias[cc] : 0.0f;
            int cr = row0 + wm * 32 + s2 * 16 + hi * 8;
#pragma unroll
            for (int v = 0; v < 8; ++v)
                C[(size_t)(cr + v) * ldc + cc] = acc[s2][j][v] + bv;
        }
}

// ---------------------------------------------------------------------------
// Weight prep: plain f32 -> bf16 convert (layout preserved, [N][K])
// ---------------------------------------------------------------------------
__global__ void convert_to_bf16(const float* __restrict__ src,
                                u16* __restrict__ dst, int count)
{
    int idx = blockIdx.x * blockDim.x + threadIdx.x;
    if (idx >= count) return;
    dst[idx] = f2bf(src[idx]);
}

// ---------------------------------------------------------------------------
// Embedding gathers (straight to bf16)
// ---------------------------------------------------------------------------
__global__ void gather_word_emb(const int* __restrict__ words,
                                const float* __restrict__ emb,
                                u16* __restrict__ x)          // [8192][256]
{
    size_t idx = (size_t)blockIdx.x * blockDim.x + threadIdx.x;
    int e = (int)(idx & (E_ - 1));
    int n = (int)(idx >> 8);
    x[idx] = f2bf(emb[(size_t)words[n] * E_ + e]);
}

// timestep-major: xc[w][m][e]
__global__ void gather_char_emb(const int* __restrict__ chars,
                                const float* __restrict__ emb,
                                u16* __restrict__ xc)         // [16][8192][256]
{
    size_t idx = (size_t)blockIdx.x * blockDim.x + threadIdx.x;
    int e = (int)(idx & (E_ - 1));
    size_t t2 = idx >> 8;
    int m = (int)(t2 & (BS_ - 1));
    int w = (int)(t2 >> 13);                       // BS_ == 2^13
    int ch = chars[(size_t)m * W_ + w];
    xc[idx] = f2bf(emb[(size_t)ch * E_ + e]);
}

// ---------------------------------------------------------------------------
// _rev_pad on a bf16 [B][S][D] buffer
// ---------------------------------------------------------------------------
__global__ void reverse_pad_u16(const u16* __restrict__ x, u16* __restrict__ y,
                                const int* __restrict__ lens, int D)
{
    size_t idx = (size_t)blockIdx.x * blockDim.x + threadIdx.x;
    if (idx >= (size_t)BS_ * D) return;
    int d  = (int)(idx % D);
    int ns = (int)(idx / D);
    int s  = ns & (S_ - 1);
    int b  = ns >> 7;
    int len = lens[b];
    int src = (s < len) ? (len - 1 - s) : s;
    y[idx] = x[((size_t)b * S_ + src) * D + d];
}

// ---------------------------------------------------------------------------
// Word-path LSTM cell.  z_t = XW[b*S+t] + zr[b]; gate layout [i,f,g,o].
// Writes output at position len-1-t for dir==1 (== _rev_pad of ys).
// ---------------------------------------------------------------------------
__global__ void lstm_cell_word(const float* __restrict__ zr,   // [64][2048]
                               const float* __restrict__ xw,   // [8192][2048]
                               float* __restrict__ h, float* __restrict__ c,
                               u16* __restrict__ hbf,          // [64][512]
                               const int* __restrict__ lens,   // [64]
                               int t, int dir,
                               u16* __restrict__ out_bf,       // [8192][1024] or null
                               float* __restrict__ out_f32,    // [8192][1024] or null
                               int outOff)
{
    int idx = blockIdx.x * blockDim.x + threadIdx.x;
    if (idx >= B_ * H_) return;
    int b = idx >> 9, u = idx & (H_ - 1);
    int len = lens[b];
    size_t zb = (size_t)b * (4 * H_);
    size_t xb = ((size_t)b * S_ + t) * (4 * H_);
    float zi = xw[xb + u]           + zr[zb + u];
    float zf = xw[xb + H_ + u]      + zr[zb + H_ + u];
    float zg = xw[xb + 2 * H_ + u]  + zr[zb + 2 * H_ + u];
    float zo = xw[xb + 3 * H_ + u]  + zr[zb + 3 * H_ + u];
    float iv = sigm(zi), fv = sigm(zf), gv = tanhf(zg), ov = sigm(zo);
    float cold = c[idx], hold = h[idx];
    float cn = fv * cold + iv * gv;
    float hn = ov * tanhf(cn);
    bool m = t < len;
    float hv = m ? hn : hold, cv = m ? cn : cold;
    h[idx] = hv; c[idx] = cv; hbf[idx] = f2bf(hv);
    int s_out = m ? (dir ? (len - 1 - t) : t) : t;
    float yv = m ? hn : 0.0f;
    size_t o = ((size_t)b * S_ + s_out) * (2 * H_) + outOff + u;
    if (out_bf)  out_bf[o]  = f2bf(yv);
    if (out_f32) out_f32[o] = yv;
}

// ---------------------------------------------------------------------------
// Char-path LSTM cell (8192 sequences, hidden 1024, gates 4096)
// ---------------------------------------------------------------------------
__global__ void lstm_cell_char(const float* __restrict__ z,    // [8192][4096]
                               float* __restrict__ h, float* __restrict__ c,
                               u16* __restrict__ hbf,          // [8192][1024]
                               const int* __restrict__ lens,   // [8192]
                               int t)
{
    size_t idx = (size_t)blockIdx.x * blockDim.x + threadIdx.x;
    if (idx >= (size_t)BS_ * 1024) return;
    int n = (int)(idx >> 10), u = (int)(idx & 1023);
    size_t zb = (size_t)n * 4096;
    float zi = z[zb + u], zf = z[zb + 1024 + u];
    float zg = z[zb + 2048 + u], zo = z[zb + 3072 + u];
    float iv = sigm(zi), fv = sigm(zf), gv = tanhf(zg), ov = sigm(zo);
    float cold = c[idx], hold = h[idx];
    float cn = fv * cold + iv * gv;
    float hn = ov * tanhf(cn);
    bool m = t < lens[n];
    float hv = m ? hn : hold, cv = m ? cn : cold;
    h[idx] = hv; c[idx] = cv; hbf[idx] = f2bf(hv);
}

// ---------------------------------------------------------------------------
// out[n][cls] = (hword[n] + valid*hc[n]) . lin_W[cls] + lin_b[cls]
// ---------------------------------------------------------------------------
__global__ void final_linear(const float* __restrict__ hword,
                             const float* __restrict__ hc,
                             const int* __restrict__ words_num,
                             const float* __restrict__ linW,   // [20][1024]
                             const float* __restrict__ linb,
                             float* __restrict__ out)
{
    int gid = blockIdx.x * blockDim.x + threadIdx.x;
    if (gid >= BS_ * NC_) return;
    int n = gid / NC_, cls = gid % NC_;
    int b = n >> 7, s = n & (S_ - 1);
    bool valid = s < words_num[b];
    const float* hw   = hword + (size_t)n * 1024;
    const float* hcp  = hc    + (size_t)n * 1024;
    const float* wrow = linW  + (size_t)cls * 1024;
    float a = linb[cls];
    for (int u = 0; u < 1024; ++u) {
        float v = hw[u] + (valid ? hcp[u] : 0.0f);
        a = fmaf(v, wrow[u], a);
    }
    out[gid] = a;
}

// ---------------------------------------------------------------------------
static void launch_gemm2(const u16* A0, int lda0, const u16* B0, int K0,
                         const u16* A1, int lda1, const u16* B1, int K1,
                         float* C, int ldc, const float* bias,
                         int M, int N, int accum, hipStream_t stream)
{
    dim3 grid((unsigned)(N / 128), (unsigned)(M / 64));
    gemm2_bf16_wmma<<<grid, 256, 0, stream>>>(A0, lda0, B0, K0, A1, lda1, B1, K1,
                                              C, ldc, bias, accum);
}

extern "C" void kernel_launch(void* const* d_in, const int* in_sizes, int n_in,
                              void* d_out, int out_size, void* d_ws, size_t ws_size,
                              hipStream_t stream)
{
    (void)in_sizes; (void)n_in; (void)out_size; (void)ws_size;

    const int*   words     = (const int*)  d_in[0];
    const int*   words_num = (const int*)  d_in[1];
    const int*   chars     = (const int*)  d_in[2];
    const int*   char_lens = (const int*)  d_in[3];
    const float* word_emb  = (const float*)d_in[4];
    const float* char_emb  = (const float*)d_in[5];
    const float* l0_Wih    = (const float*)d_in[6];
    const float* l0_Whh    = (const float*)d_in[7];
    const float* l0_b      = (const float*)d_in[8];
    const float* l1_Wih    = (const float*)d_in[9];
    const float* l1_Whh    = (const float*)d_in[10];
    const float* l1_b      = (const float*)d_in[11];
    const float* c_Wih     = (const float*)d_in[12];
    const float* c_Whh     = (const float*)d_in[13];
    const float* c_b       = (const float*)d_in[14];
    const float* lin_W     = (const float*)d_in[15];
    const float* lin_b     = (const float*)d_in[16];
    float* out = (float*)d_out;

    // ---- workspace carve-up (256B aligned) ----
    size_t off = 0;
    auto take = [&](size_t bytes) -> char* {
        char* p = (char*)d_ws + off;
        off += (bytes + 255) & ~(size_t)255;
        return p;
    };
    u16* wt_l0_ih[2], *wt_l0_hh[2], *wt_l1_ih[2], *wt_l1_hh[2];
    for (int d = 0; d < 2; ++d) {
        wt_l0_ih[d] = (u16*)take((size_t)2048 * 256  * 2);
        wt_l0_hh[d] = (u16*)take((size_t)2048 * 512  * 2);
        wt_l1_ih[d] = (u16*)take((size_t)2048 * 1024 * 2);
        wt_l1_hh[d] = (u16*)take((size_t)2048 * 512  * 2);
    }
    u16*   wt_c_ih = (u16*)take((size_t)4096 * 256  * 2);
    u16*   wt_c_hh = (u16*)take((size_t)4096 * 1024 * 2);
    u16*   x0    = (u16*)  take((size_t)BS_ * E_  * 2);
    u16*   x0r   = (u16*)  take((size_t)BS_ * E_  * 2);
    u16*   h1    = (u16*)  take((size_t)BS_ * 1024 * 2);
    u16*   h1r   = (u16*)  take((size_t)BS_ * 1024 * 2);
    float* hword = (float*)take((size_t)BS_ * 1024 * 4);
    float* XW    = (float*)take((size_t)BS_ * 2048 * 4);
    float* zr    = (float*)take((size_t)64  * 2048 * 4);
    float* hst   = (float*)take((size_t)64  * 512  * 4);
    float* cst   = (float*)take((size_t)64  * 512  * 4);
    u16*   hbf   = (u16*)  take((size_t)64  * 512  * 2);
    u16*   xc    = (u16*)  take((size_t)NBS_ * E_  * 2);
    float* zc    = (float*)take((size_t)BS_ * 4096 * 4);
    float* hc    = (float*)take((size_t)BS_ * 1024 * 4);
    float* cc    = (float*)take((size_t)BS_ * 1024 * 4);
    u16*   hcbf  = (u16*)  take((size_t)BS_ * 1024 * 2);

    // ---- Phase A: weights -> bf16 (layout preserved, already B-friendly) ----
    auto cvt = [&](const float* src, u16* dst, int count) {
        convert_to_bf16<<<dim3((unsigned)((count + 255) / 256)), 256, 0, stream>>>(
            src, dst, count);
    };
    for (int d = 0; d < 2; ++d) {
        cvt(l0_Wih + (size_t)d * 2048 * 256,  wt_l0_ih[d], 2048 * 256);
        cvt(l0_Whh + (size_t)d * 2048 * 512,  wt_l0_hh[d], 2048 * 512);
        cvt(l1_Wih + (size_t)d * 2048 * 1024, wt_l1_ih[d], 2048 * 1024);
        cvt(l1_Whh + (size_t)d * 2048 * 512,  wt_l1_hh[d], 2048 * 512);
    }
    cvt(c_Wih, wt_c_ih, 4096 * 256);
    cvt(c_Whh, wt_c_hh, 4096 * 1024);

    // ---- Phase B: word path ----
    gather_word_emb<<<dim3((unsigned)((size_t)BS_ * E_ / 256)), 256, 0, stream>>>(
        words, word_emb, x0);
    reverse_pad_u16<<<dim3((unsigned)((size_t)BS_ * E_ / 256)), 256, 0, stream>>>(
        x0, x0r, words_num, E_);

    auto run_word_layer = [&](int layer) {
        for (int d = 0; d < 2; ++d) {
            const u16* X; int K; const u16* wih; const u16* whh; const float* bias;
            if (layer == 0) { X = d ? x0r : x0; K = 256;
                              wih = wt_l0_ih[d]; whh = wt_l0_hh[d];
                              bias = l0_b + (size_t)d * 2048; }
            else            { X = d ? h1r : h1; K = 1024;
                              wih = wt_l1_ih[d]; whh = wt_l1_hh[d];
                              bias = l1_b + (size_t)d * 2048; }
            // hoisted input projection for all timesteps: XW = X @ Wih^T + b
            launch_gemm2(X, K, wih, K, nullptr, 0, nullptr, 0,
                         XW, 2048, bias, BS_, 2048, 0, stream);
            (void)hipMemsetAsync(hst, 0, (size_t)64 * 512 * 4, stream);
            (void)hipMemsetAsync(cst, 0, (size_t)64 * 512 * 4, stream);
            (void)hipMemsetAsync(hbf, 0, (size_t)64 * 512 * 2, stream);
            for (int t = 0; t < S_; ++t) {
                launch_gemm2(hbf, 512, whh, 512, nullptr, 0, nullptr, 0,
                             zr, 2048, nullptr, 64, 2048, 0, stream);
                lstm_cell_word<<<dim3(64 * 512 / 256), 256, 0, stream>>>(
                    zr, XW, hst, cst, hbf, words_num, t, d,
                    layer == 0 ? h1 : nullptr,
                    layer == 0 ? nullptr : hword,
                    d * 512);
            }
        }
    };
    run_word_layer(0);
    reverse_pad_u16<<<dim3((unsigned)((size_t)BS_ * 1024 / 256)), 256, 0, stream>>>(
        h1, h1r, words_num, 1024);
    run_word_layer(1);

    // ---- Phase C: char path (the ~1.4 TFLOP hot loop) ----
    gather_char_emb<<<dim3((unsigned)((size_t)NBS_ * E_ / 256)), 256, 0, stream>>>(
        chars, char_emb, xc);
    (void)hipMemsetAsync(hc,   0, (size_t)BS_ * 1024 * 4, stream);
    (void)hipMemsetAsync(cc,   0, (size_t)BS_ * 1024 * 4, stream);
    (void)hipMemsetAsync(hcbf, 0, (size_t)BS_ * 1024 * 2, stream);
    for (int t = 0; t < W_; ++t) {
        // z = x_t @ cWih^T + h @ cWhh^T + c_b   (single fused kernel: no zc
        // round-trip between the two matmuls)
        launch_gemm2(xc + (size_t)t * BS_ * E_, E_, wt_c_ih, E_,
                     hcbf, 1024, wt_c_hh, 1024,
                     zc, 4096, c_b, BS_, 4096, 0, stream);
        lstm_cell_char<<<dim3((unsigned)((size_t)BS_ * 1024 / 256)), 256, 0, stream>>>(
            zc, hc, cc, hcbf, char_lens, t);
    }

    // ---- Phase D: (hword + masked hc) @ lin_W^T + lin_b ----
    final_linear<<<dim3((unsigned)((BS_ * NC_ + 255) / 256)), 256, 0, stream>>>(
        hword, hc, words_num, lin_W, lin_b, out);
}
